// EncoderBlock_67912022884724
// MI455X (gfx1250) — matrix operand order repaired and checked by reading, not compiled, not run on hardware
//
#include <hip/hip_runtime.h>

typedef __bf16 bf16;
typedef __attribute__((ext_vector_type(16))) __bf16 v16bf;
typedef __attribute__((ext_vector_type(8)))  float  v8f;

#define SEQ   2048
#define LOG2E 1.4426950408889634f
#define BK    64

union FragU { uint4 u[2]; v16bf v; bf16 e[16]; };

__device__ inline v8f v8zero(){ v8f z; for (int i = 0; i < 8; ++i) z[i] = 0.f; return z; }

// A-matrix 16x32 bf16 fragment (ISA 7.12.2): lanes 0-15 rows M=0..15, K-chunks {kg*8..+7, kg*8+16..+23}
__device__ inline v16bf load_frag_a(const bf16* p, int strideBF, int row, int kbase, int kg){
  FragU f;
  const bf16* base = p + (size_t)row * strideBF + kbase + (kg << 3);
  f.u[0] = *reinterpret_cast<const uint4*>(base);
  f.u[1] = *reinterpret_cast<const uint4*>(base + 16);
  return f.v;
}

// B-matrix 32x16 bf16 fragment from K-major (transposed) storage: K-chunks {kg*8..+7, kg*8+16..+23}
__device__ inline v16bf load_frag_b(const bf16* p, int strideBF, int col, int kbase, int kg){
  FragU f;
  const bf16* base = p + (size_t)col * strideBF + kbase + (kg << 3);
  f.u[0] = *reinterpret_cast<const uint4*>(base);
  f.u[1] = *reinterpret_cast<const uint4*>(base + 16);
  return f.v;
}

// ---- CDNA5 async global->LDS copies (bypass VGPRs, tracked by ASYNCcnt) ----
__device__ inline void async_copy_32B(unsigned lds, const bf16* g){
  unsigned long long ga = (unsigned long long)(uintptr_t)g;
  asm volatile("global_load_async_to_lds_b128 %0, %1, off\n\t"
               "global_load_async_to_lds_b128 %0, %1, off offset:16"
               :: "v"(lds), "v"(ga) : "memory");
}
__device__ inline void async_copy_64B(unsigned lds, const bf16* g){
  unsigned long long ga = (unsigned long long)(uintptr_t)g;
  asm volatile("global_load_async_to_lds_b128 %0, %1, off\n\t"
               "global_load_async_to_lds_b128 %0, %1, off offset:16\n\t"
               "global_load_async_to_lds_b128 %0, %1, off offset:32\n\t"
               "global_load_async_to_lds_b128 %0, %1, off offset:48"
               :: "v"(lds), "v"(ga) : "memory");
}
__device__ inline void wait_async0(){
  asm volatile("s_wait_asynccnt 0x0" ::: "memory");
}

// ---------------- weight convert + transpose: W[K][N] f32 -> Wt[N][K] bf16 ----------------
__global__ void convt_kernel(const float* __restrict__ in, bf16* __restrict__ out, int K, int N){
  int idx = blockIdx.x * 256 + threadIdx.x;
  int k = idx / N, n = idx - k * N;
  out[(size_t)n * K + k] = (bf16)in[idx];
}

// ---------------- layernorm over rows of 512, f32 -> bf16 (eps added to std, like ref) ----------------
__global__ void __launch_bounds__(256) ln_kernel(const float* __restrict__ x,
                                                 const float* __restrict__ gamma,
                                                 const float* __restrict__ beta,
                                                 bf16* __restrict__ out){
  __shared__ float red[256];
  int row = blockIdx.x, t = threadIdx.x;
  const float* xr = x + (size_t)row * 512;
  float a = xr[t], b = xr[t + 256];
  red[t] = a + b; __syncthreads();
  for (int o = 128; o > 0; o >>= 1){ if (t < o) red[t] += red[t + o]; __syncthreads(); }
  float mean = red[0] * (1.f / 512.f);
  __syncthreads();
  float da = a - mean, db = b - mean;
  red[t] = da * da + db * db; __syncthreads();
  for (int o = 128; o > 0; o >>= 1){ if (t < o) red[t] += red[t + o]; __syncthreads(); }
  float inv = 1.f / (sqrtf(red[0] * (1.f / 512.f)) + 1e-6f);
  bf16* orow = out + (size_t)row * 512;
  orow[t]       = (bf16)(gamma[t] * da * inv + beta[t]);
  orow[t + 256] = (bf16)(gamma[t + 256] * db * inv + beta[t + 256]);
}

// ---------------- GEMM: out[M,N] = A[M,K](bf16) * Wt[N,K](bf16)^T + bias (+relu) (+residual) ----------------
// block tile 128x128, 8 waves (4x2 grid (M x N) of 32x64 wave tiles), K-step 64,
// double-buffered LDS fed by async global->LDS DMA overlapped with WMMA.
__global__ void __launch_bounds__(256) gemm_kernel(
    const bf16* __restrict__ A, const bf16* __restrict__ Wt,
    const float* __restrict__ bias, const float* __restrict__ residual,
    float* __restrict__ outF, bf16* __restrict__ outB,
    int N, int K, int relu){
  __shared__ bf16 As[2][128 * BK];
  __shared__ bf16 Bs[2][128 * BK];
  int m0 = blockIdx.x * 128, n0 = blockIdx.y * 128;
  int tid = threadIdx.x;
  int w = tid >> 5, lane = tid & 31, l16 = lane & 15, kg = lane >> 4;
  int wm = w >> 1, wn = w & 1;
  int lr = tid >> 1;            // staging row (0..127)
  int lc = (tid & 1) << 5;      // staging col base (0 or 32 elements)

  const bf16* gA = A  + (size_t)(m0 + lr) * K + lc;
  const bf16* gB = Wt + (size_t)(n0 + lr) * K + lc;
  unsigned ldsA[2], ldsB[2];
  for (int buf = 0; buf < 2; ++buf){
    ldsA[buf] = (unsigned)(uintptr_t)(const void*)(&As[buf][lr * BK + lc]);
    ldsB[buf] = (unsigned)(uintptr_t)(const void*)(&Bs[buf][lr * BK + lc]);
  }

  v8f acc[2][4];
  for (int i = 0; i < 2; ++i) for (int j = 0; j < 4; ++j) acc[i][j] = v8zero();

  int nT = K / BK;
  async_copy_64B(ldsA[0], gA);
  async_copy_64B(ldsB[0], gB);

  for (int t = 0; t < nT; ++t){
    int buf = t & 1;
    wait_async0();
    __syncthreads();
    if (t + 1 < nT){
      async_copy_64B(ldsA[buf ^ 1], gA + (t + 1) * BK);
      async_copy_64B(ldsB[buf ^ 1], gB + (t + 1) * BK);
    }
    const bf16* as = As[buf];
    const bf16* bs = Bs[buf];
    v16bf af[2][2], bfr[2][4];
    for (int kk = 0; kk < 2; ++kk){
      for (int mt = 0; mt < 2; ++mt) af[kk][mt]  = load_frag_a(as, BK, wm * 32 + mt * 16 + l16, kk * 32, kg);
      for (int nt = 0; nt < 4; ++nt) bfr[kk][nt] = load_frag_b(bs, BK, wn * 64 + nt * 16 + l16, kk * 32, kg);
    }
    for (int kk = 0; kk < 2; ++kk)
      for (int mt = 0; mt < 2; ++mt)
        for (int nt = 0; nt < 4; ++nt)
          acc[mt][nt] = __builtin_amdgcn_wmma_f32_16x16x32_bf16(
              false, af[kk][mt], false, bfr[kk][nt], (short)0, acc[mt][nt], false, false);
    __syncthreads();
  }

  for (int mt = 0; mt < 2; ++mt){
    for (int nt = 0; nt < 4; ++nt){
      int col = n0 + wn * 64 + nt * 16 + l16;
      float bv = bias[col];
      for (int r = 0; r < 8; ++r){
        int row = m0 + wm * 32 + mt * 16 + r + (kg << 3);
        float v = acc[mt][nt][r] + bv;
        if (relu) v = fmaxf(v, 0.f);
        size_t idx = (size_t)row * N + col;
        if (residual) v += residual[idx];
        if (outB) outB[idx] = (bf16)v;
        else      outF[idx] = v;
      }
    }
  }
}

// ---------------- flash attention: 128 queries/block, 64-key tiles, online softmax ----------------
__global__ void __launch_bounds__(256) attn_kernel(
    const bf16* __restrict__ Q, const bf16* __restrict__ Kb, const bf16* __restrict__ Vb,
    const int* __restrict__ mask, bf16* __restrict__ ctx){
  __shared__ bf16 Ks[64 * 64];      // [key][dh]  (natural => B-frag for Q.K^T), async-staged
  __shared__ bf16 Vt[64 * 64];      // [dh][key]  (transposed => B-frag for P.V)
  __shared__ bf16 Ps[8 * 16 * 64];  // per-wave P scratch (C-layout -> A-layout)
  int q0 = blockIdx.x * 128;
  int bh = blockIdx.y;
  int b = bh >> 3, h = bh & 7;
  int tid = threadIdx.x;
  int w = tid >> 5, lane = tid & 31, l16 = lane & 15, kg = lane >> 4;

  v16bf qf[2];
  for (int kt = 0; kt < 2; ++kt)
    qf[kt] = load_frag_a(Q, 512, b * SEQ + q0 + w * 16 + l16, h * 64 + kt * 32, kg);

  v8f o[4]; for (int i = 0; i < 4; ++i) o[i] = v8zero();
  float mrow[8], lrow[8];
  for (int r = 0; r < 8; ++r){ mrow[r] = -__builtin_inff(); lrow[r] = 0.f; }

  int lr = tid >> 2, lc = (tid & 3) << 4;
  unsigned ldsK = (unsigned)(uintptr_t)(const void*)(&Ks[lr * 64 + lc]);
  bf16* pw = Ps + w * 16 * 64;

  for (int kt0 = 0; kt0 < SEQ / 64; ++kt0){
    {
      int key = kt0 * 64 + lr;
      const bf16* gk = Kb + (size_t)(b * SEQ + key) * 512 + h * 64 + lc;
      async_copy_32B(ldsK, gk);                      // K tile: DMA straight to LDS
      const uint4* gv = reinterpret_cast<const uint4*>(Vb + (size_t)(b * SEQ + key) * 512 + h * 64 + lc);
      FragU tv; tv.u[0] = gv[0]; tv.u[1] = gv[1];    // V tile: in-register transpose
      for (int i = 0; i < 16; ++i) Vt[(lc + i) * 64 + lr] = tv.e[i];
    }
    wait_async0();
    __syncthreads();

    // scores: 16 queries x 64 keys per wave
    float sc[4][8];
    for (int nt = 0; nt < 4; ++nt){
      v8f a = v8zero();
      for (int kt = 0; kt < 2; ++kt){
        v16bf kf = load_frag_b(Ks, 64, nt * 16 + l16, kt * 32, kg);
        a = __builtin_amdgcn_wmma_f32_16x16x32_bf16(false, qf[kt], false, kf, (short)0, a, false, false);
      }
      int key = kt0 * 64 + nt * 16 + l16;
      int mv = mask[b * SEQ + key];
      for (int r = 0; r < 8; ++r)
        sc[nt][r] = (mv == 0) ? -1e9f : a[r] * 0.125f;  // 1/sqrt(64)
    }

    // online softmax: row-wise max/sum via 16-lane butterflies (lanes = key columns)
    float corr[8];
    for (int r = 0; r < 8; ++r){
      float tm = fmaxf(fmaxf(sc[0][r], sc[1][r]), fmaxf(sc[2][r], sc[3][r]));
      for (int off = 8; off > 0; off >>= 1) tm = fmaxf(tm, __shfl_xor(tm, off, 32));
      float mn = fmaxf(mrow[r], tm);
      corr[r] = exp2f((mrow[r] - mn) * LOG2E);
      mrow[r] = mn;
    }
    for (int r = 0; r < 8; ++r){
      float rs = 0.f;
      for (int nt = 0; nt < 4; ++nt){
        float p = exp2f((sc[nt][r] - mrow[r]) * LOG2E);
        sc[nt][r] = p; rs += p;
      }
      for (int off = 8; off > 0; off >>= 1) rs += __shfl_xor(rs, off, 32);
      lrow[r] = lrow[r] * corr[r] + rs;
    }
    for (int nt = 0; nt < 4; ++nt)
      for (int r = 0; r < 8; ++r)
        o[nt][r] *= corr[r];

    // P (C-layout) -> per-wave LDS -> A-frags (same-wave LDS is in-order; no barrier needed)
    for (int nt = 0; nt < 4; ++nt)
      for (int r = 0; r < 8; ++r)
        pw[(r + (kg << 3)) * 64 + nt * 16 + l16] = (bf16)sc[nt][r];

    for (int kt = 0; kt < 2; ++kt){
      v16bf pf = load_frag_a(pw, 64, l16, kt * 32, kg);
      for (int nt = 0; nt < 4; ++nt){
        v16bf vf = load_frag_b(Vt, 64, nt * 16 + l16, kt * 32, kg);
        o[nt] = __builtin_amdgcn_wmma_f32_16x16x32_bf16(false, pf, false, vf, (short)0, o[nt], false, false);
      }
    }
    __syncthreads();
  }

  for (int nt = 0; nt < 4; ++nt){
    int dh = nt * 16 + l16;
    for (int r = 0; r < 8; ++r){
      int q = q0 + w * 16 + r + (kg << 3);
      float v = o[nt][r] / lrow[r];
      ctx[(size_t)(b * SEQ + q) * 512 + h * 64 + dh] = (bf16)v;
    }
  }
}

// ---------------- host-side orchestration ----------------
extern "C" void kernel_launch(void* const* d_in, const int* in_sizes, int n_in,
                              void* d_out, int out_size, void* d_ws, size_t ws_size,
                              hipStream_t stream){
  const float* x    = (const float*)d_in[0];
  const int*   mask = (const int*)  d_in[1];
  const float* Wq = (const float*)d_in[2];  const float* bq = (const float*)d_in[3];
  const float* Wk = (const float*)d_in[4];  const float* bk = (const float*)d_in[5];
  const float* Wv = (const float*)d_in[6];  const float* bv = (const float*)d_in[7];
  const float* Wo = (const float*)d_in[8];  const float* bo = (const float*)d_in[9];
  const float* W1 = (const float*)d_in[10]; const float* b1 = (const float*)d_in[11];
  const float* W2 = (const float*)d_in[12]; const float* b2 = (const float*)d_in[13];
  const float* gamma = (const float*)d_in[14];
  const float* beta  = (const float*)d_in[15];
  float* out = (float*)d_out;

  char* ws = (char*)d_ws;
  size_t off = 0;
  auto alloc = [&](size_t bytes){ void* p = ws + off; off += bytes; return p; };
  bf16*  wqT = (bf16*)alloc((size_t)512 * 512 * 2);
  bf16*  wkT = (bf16*)alloc((size_t)512 * 512 * 2);
  bf16*  wvT = (bf16*)alloc((size_t)512 * 512 * 2);
  bf16*  woT = (bf16*)alloc((size_t)512 * 512 * 2);
  bf16*  w1T = (bf16*)alloc((size_t)512 * 2048 * 2);
  bf16*  w2T = (bf16*)alloc((size_t)2048 * 512 * 2);
  bf16*  h   = (bf16*)alloc((size_t)8192 * 512 * 2);
  float* x1  = (float*)alloc((size_t)8192 * 512 * 4);
  bf16*  qb  = (bf16*)alloc((size_t)8192 * 512 * 2);
  bf16*  kb  = (bf16*)alloc((size_t)8192 * 512 * 2);
  bf16*  vb  = (bf16*)alloc((size_t)8192 * 512 * 2);
  bf16*  cb  = (bf16*)alloc((size_t)8192 * 512 * 2);
  bf16*  f1  = qb;  // alias: f1 [8192,2048] reuses dead q/k/v/ctx region (contiguous 32 MB)

  // weights -> bf16, transposed [N][K]
  convt_kernel<<<512 * 512 / 256, 256, 0, stream>>>(Wq, wqT, 512, 512);
  convt_kernel<<<512 * 512 / 256, 256, 0, stream>>>(Wk, wkT, 512, 512);
  convt_kernel<<<512 * 512 / 256, 256, 0, stream>>>(Wv, wvT, 512, 512);
  convt_kernel<<<512 * 512 / 256, 256, 0, stream>>>(Wo, woT, 512, 512);
  convt_kernel<<<512 * 2048 / 256, 256, 0, stream>>>(W1, w1T, 512, 2048);
  convt_kernel<<<2048 * 512 / 256, 256, 0, stream>>>(W2, w2T, 2048, 512);

  // h = LN(x)
  ln_kernel<<<8192, 256, 0, stream>>>(x, gamma, beta, h);

  dim3 g512(8192 / 128, 512 / 128);
  gemm_kernel<<<g512, 256, 0, stream>>>(h, wqT, bq, nullptr, nullptr, qb, 512, 512, 0);
  gemm_kernel<<<g512, 256, 0, stream>>>(h, wkT, bk, nullptr, nullptr, kb, 512, 512, 0);
  gemm_kernel<<<g512, 256, 0, stream>>>(h, wvT, bv, nullptr, nullptr, vb, 512, 512, 0);

  dim3 gat(SEQ / 128, 4 * 8);
  attn_kernel<<<gat, 256, 0, stream>>>(qb, kb, vb, mask, cb);

  // x1 = x + ctx @ Wo + bo
  gemm_kernel<<<g512, 256, 0, stream>>>(cb, woT, bo, x, x1, nullptr, 512, 512, 0);

  // h = LN(x1)  (same gamma/beta)
  ln_kernel<<<8192, 256, 0, stream>>>(x1, gamma, beta, h);

  // f1 = relu(h @ W1 + b1)
  dim3 g2048(8192 / 128, 2048 / 128);
  gemm_kernel<<<g2048, 256, 0, stream>>>(h, w1T, b1, nullptr, nullptr, f1, 2048, 512, 1);

  // out = x1 + f1 @ W2 + b2
  gemm_kernel<<<g512, 256, 0, stream>>>(f1, w2T, b2, x1, out, nullptr, 512, 2048, 0);
}
